// MiCTRANet_78065325572262
// MI455X (gfx1250) — matrix operand order repaired and checked by reading, not compiled, not run on hardware
//
#include <hip/hip_runtime.h>
#include <hip/hip_bf16.h>

typedef __attribute__((ext_vector_type(16))) _Float16 v16h;
typedef __attribute__((ext_vector_type(8)))  _Float16 v8h;
typedef __attribute__((ext_vector_type(8)))  float    v8f;

#define L_DIM 128
#define N_DIM 49
#define B_DIM 16
#define C_DIM 512
#define H_DIM 512
#define O_DIM 40
#define H4_DIM 2048

// ---------------------------------------------------------------- WMMA utils

__device__ __forceinline__ v8f wmma16(v16h a, v16h b, v8f c) {
  // D = A(16x32 f16) * B(32x16 f16) + C(16x16 f32)
  return __builtin_amdgcn_wmma_f32_16x16x32_f16(
      /*neg_a=*/false, a, /*neg_b=*/false, b,
      /*c_mod=*/(short)0, c, /*reuse_a=*/false, /*reuse_b=*/false);
}

// A-matrix fragment from a row-major f16 tile (16 x ldk), rows are M.
// Layout (ISA 7.12.2, 16-bit A 16x32): lane&15 = M row; lanes<16 hold
// K = kb+0..7 (halves 0..7) and kb+16..23 (halves 8..15); lanes>=16 hold
// K = kb+8..15 and kb+24..31.
__device__ __forceinline__ v16h afrag_lds(const _Float16* base, int ldk,
                                          int kb, int lane) {
  int row = lane & 15;
  int hi  = (lane >> 4) & 1;
  const _Float16* p0 = base + row * ldk + kb + hi * 8;
  const _Float16* p1 = base + row * ldk + kb + 16 + hi * 8;
  v8h lo = *(const v8h*)p0;
  v8h hh = *(const v8h*)p1;
  v16h a;
#pragma unroll
  for (int i = 0; i < 8; ++i) { a[i] = lo[i]; a[8 + i] = hh[i]; }
  return a;
}

// B-matrix fragment from an N-major f16 weight (row n contiguous over K).
// Layout (ISA 7.12.2, 16-bit B 32x16): lane&15 = N column; lanes<16 hold
// K = kb+0..7 and kb+16..23, lanes>=16 hold K = kb+8..15 and kb+24..31.
__device__ __forceinline__ v16h bfrag_g(const _Float16* __restrict__ W,
                                        int ldk, int nbase, int kb, int lane) {
  int col = nbase + (lane & 15);
  int k0  = kb + ((lane >> 4) & 1) * 8;
  const _Float16* p = W + (size_t)col * ldk + k0;
  v8h lo = *(const v8h*)p;
  v8h hh = *(const v8h*)(p + 16);
  v16h b;
#pragma unroll
  for (int i = 0; i < 8; ++i) { b[i] = lo[i]; b[8 + i] = hh[i]; }
  return b;
}

__device__ __forceinline__ float sigmoidf(float x) {
  return 1.f / (1.f + __expf(-x));
}

// ------------------------------------------------------------- prep kernels

// in: (K,N) f32 row-major -> out: (N,K) f16 row-major
__global__ __launch_bounds__(256) void k0_transpose(
    const float* __restrict__ in, _Float16* __restrict__ out, int K, int N) {
  int idx = blockIdx.x * 256 + threadIdx.x;
  if (idx >= K * N) return;
  int k = idx / N, n = idx % N;
  out[(size_t)n * K + k] = (_Float16)in[idx];
}

__global__ __launch_bounds__(256) void k0_convert(
    const float* __restrict__ in, _Float16* __restrict__ out, int n) {
  int idx = blockIdx.x * 256 + threadIdx.x;
  if (idx < n) out[idx] = (_Float16)in[idx];
}

// --------------------------------------------------- key/val big GEMM (f16)
// key[m,n] = sum_k feat[m,k] * Wk[k,n]   (M = L*N*B = 100352, K = N = 512)
// One block (512 thr, 16 waves) = one 16-row M tile staged to LDS; each wave
// owns 32 output columns for BOTH key and val (4 accumulators). The k-loop is
// explicitly software-pipelined: the NEXT k-step's A/B fragments are loaded
// before the CURRENT k-step's WMMAs consume their fragments, so the
// s_wait_loadcnt for in-flight loads lands after the matrix ops and global
// load latency overlaps WMMA issue within a single wave.

__global__ __launch_bounds__(512, 1) void k1_keyval(
    const float* __restrict__ feat,
    const _Float16* __restrict__ WkT,   // (H, C) f16, N-major
    const _Float16* __restrict__ WvT,   // (H, C) f16, N-major
    _Float16* __restrict__ key16,
    _Float16* __restrict__ val16) {
  __shared__ __align__(16) _Float16 aT[16 * C_DIM];
  size_t m0 = (size_t)blockIdx.x * 16;

  const float4* src4 = (const float4*)(feat + m0 * C_DIM);
  for (int i = threadIdx.x; i < 16 * C_DIM / 4; i += 512) {
    float4 f = src4[i];
    int b4 = i * 4;
    aT[b4 + 0] = (_Float16)f.x; aT[b4 + 1] = (_Float16)f.y;
    aT[b4 + 2] = (_Float16)f.z; aT[b4 + 3] = (_Float16)f.w;
  }
  __syncthreads();

  int wave = threadIdx.x >> 5;       // 0..15
  int lane = threadIdx.x & 31;
  int nbase = wave * 32;             // 2 N-tiles of 16 per wave

  v8f aK0 = {}, aK1 = {}, aV0 = {}, aV1 = {};

  // Prologue: fragments for kb = 0.
  v16h a   = afrag_lds(aT, C_DIM, 0, lane);
  v16h bk0 = bfrag_g(WkT, C_DIM, nbase +  0, 0, lane);
  v16h bk1 = bfrag_g(WkT, C_DIM, nbase + 16, 0, lane);
  v16h bv0 = bfrag_g(WvT, C_DIM, nbase +  0, 0, lane);
  v16h bv1 = bfrag_g(WvT, C_DIM, nbase + 16, 0, lane);

  for (int kb = 0; kb < C_DIM - 32; kb += 32) {
    // Issue next k-step's loads first (no dependence on current WMMAs).
    v16h aN  = afrag_lds(aT, C_DIM, kb + 32, lane);
    v16h k0n = bfrag_g(WkT, C_DIM, nbase +  0, kb + 32, lane);
    v16h k1n = bfrag_g(WkT, C_DIM, nbase + 16, kb + 32, lane);
    v16h v0n = bfrag_g(WvT, C_DIM, nbase +  0, kb + 32, lane);
    v16h v1n = bfrag_g(WvT, C_DIM, nbase + 16, kb + 32, lane);
    // Consume current fragments.
    aK0 = wmma16(a, bk0, aK0);
    aK1 = wmma16(a, bk1, aK1);
    aV0 = wmma16(a, bv0, aV0);
    aV1 = wmma16(a, bv1, aV1);
    // Rotate buffers.
    a = aN; bk0 = k0n; bk1 = k1n; bv0 = v0n; bv1 = v1n;
  }
  // Epilogue: last k-step.
  aK0 = wmma16(a, bk0, aK0);
  aK1 = wmma16(a, bk1, aK1);
  aV0 = wmma16(a, bv0, aV0);
  aV1 = wmma16(a, bv1, aV1);

  int nlo = lane & 15;
  int mr  = (lane >> 4) * 8;
#define K1_STORE(ACC_K, ACC_V, J)                                        \
  {                                                                      \
    size_t nn = (size_t)nbase + (J) * 16 + nlo;                          \
    _Float16* kp = key16 + (m0 + mr) * H_DIM + nn;                       \
    _Float16* vp = val16 + (m0 + mr) * H_DIM + nn;                       \
    _Pragma("unroll") for (int i = 0; i < 8; ++i) {                      \
      kp[(size_t)i * H_DIM] = (_Float16)ACC_K[i];                        \
      vp[(size_t)i * H_DIM] = (_Float16)ACC_V[i];                        \
    }                                                                    \
  }
  K1_STORE(aK0, aV0, 0)
  K1_STORE(aK1, aV1, 1)
#undef K1_STORE
}

// ------------------------------------------------------- sequential scan
// Single workgroup, 32 waves. All per-step GEMMs via WMMA f16->f32.
// Next step's key/val HBM slices are prefetched (global_prefetch_b8) while
// the current step computes: a single WGP cannot otherwise cover HBM latency.

__global__ __launch_bounds__(1024) void k2_scan(
    const _Float16* __restrict__ key16,
    const _Float16* __restrict__ val16,
    const float* __restrict__ prior,
    const float* __restrict__ h0h,
    const float* __restrict__ h0c,
    const float* __restrict__ vvec,
    const _Float16* __restrict__ WhT,   // (H,H) N-major
    const _Float16* __restrict__ Wih,   // (4H,C) row-major == N-major for x@W^T
    const _Float16* __restrict__ Whh,   // (4H,H) row-major == N-major for h@W^T
    const float* __restrict__ bih,
    const float* __restrict__ bhh,
    float* __restrict__ hs,             // (L,B,H) f32
    float* __restrict__ qbuf,           // (B,H) f32 scratch
    float* __restrict__ gbuf,           // (B,4H) f32 scratch
    float* __restrict__ outh,
    float* __restrict__ outc) {
  __shared__ __align__(16) _Float16 h16[B_DIM * H_DIM];
  __shared__ __align__(16) _Float16 x16[B_DIM * H_DIM];
  __shared__ float s_vv[H_DIM];
  __shared__ float s_sc[N_DIM * B_DIM];
  __shared__ float s_aw[N_DIM * B_DIM];
  __shared__ float s_sw[B_DIM];

  int tid  = threadIdx.x;
  int wave = tid >> 5;
  int lane = tid & 31;

  float cpriv[8];
#pragma unroll
  for (int j = 0; j < 8; ++j) {
    int idx = tid + j * 1024;
    cpriv[j] = h0c[idx];
    h16[idx] = (_Float16)h0h[idx];
  }
  for (int i = tid; i < H_DIM; i += 1024) s_vv[i] = vvec[i];
  __syncthreads();

  const size_t slice = (size_t)N_DIM * B_DIM * H_DIM;   // halves per step

  for (int l = 0; l < L_DIM; ++l) {
    // --- prefetch next step's key/val slices into cache (128B per probe) ---
    if (l + 1 < L_DIM) {
      const _Float16* nk = key16 + (size_t)(l + 1) * slice;
      const _Float16* nv = val16 + (size_t)(l + 1) * slice;
      for (int i = tid; i < (int)(slice / 64); i += 1024) {
        __builtin_prefetch(nk + (size_t)i * 64, 0, 1);
        __builtin_prefetch(nv + (size_t)i * 64, 0, 1);
      }
    }

    // --- A) query = h @ Wh : wave w owns N-tile w ---
    {
      v8f q = {};
      int nb = wave * 16;
      for (int kb = 0; kb < H_DIM; kb += 32) {
        v16h a = afrag_lds(h16, H_DIM, kb, lane);
        q = wmma16(a, bfrag_g(WhT, H_DIM, nb, kb, lane), q);
      }
      int nn = nb + (lane & 15);
      int mr = (lane >> 4) * 8;
#pragma unroll
      for (int i = 0; i < 8; ++i) qbuf[(mr + i) * H_DIM + nn] = q[i];
    }
    __threadfence_block();
    __syncthreads();

    // --- B) scores[n,b] = sum_h vv[h]*tanh(q[b,h] + key[l,n,b,h]) ---
    if (tid < N_DIM * B_DIM) {
      int n = tid >> 4, b = tid & 15;
      const _Float16* kp = key16 + (((size_t)l * N_DIM + n) * B_DIM + b) * H_DIM;
      const float* qp = qbuf + b * H_DIM;
      float acc = 0.f;
      for (int h0 = 0; h0 < H_DIM; h0 += 8) {
        v8h kv = *(const v8h*)(kp + h0);
#pragma unroll
        for (int i = 0; i < 8; ++i)
          acc += s_vv[h0 + i] * tanhf(qp[h0 + i] + (float)kv[i]);
      }
      s_sc[tid] = acc;
    }
    __syncthreads();

    // --- C) softmax over n, apply prior, sw clip ---
    if (tid < B_DIM) {
      int b = tid;
      float mx = -1e30f;
      for (int n = 0; n < N_DIM; ++n) mx = fmaxf(mx, s_sc[n * B_DIM + b]);
      float sum = 0.f;
      for (int n = 0; n < N_DIM; ++n) sum += __expf(s_sc[n * B_DIM + b] - mx);
      float inv = 1.f / sum;
      float sw = 0.f;
      for (int n = 0; n < N_DIM; ++n) {
        float aw = __expf(s_sc[n * B_DIM + b] - mx) * inv *
                   prior[((size_t)l * N_DIM + n) * B_DIM + b];
        s_aw[n * B_DIM + b] = aw;
        sw += aw;
      }
      s_sw[b] = fmaxf(sw, 1e-5f);
    }
    __syncthreads();

    // --- D) context[b,c] = sum_n aw[n,b]*val[l,n,b,c]; x = context/sw ---
    {
      int b = tid >> 6, cg = tid & 63;
      const _Float16* vp =
          val16 + (((size_t)l * N_DIM) * B_DIM + b) * H_DIM + cg * 8;
      float acc[8] = {0.f, 0.f, 0.f, 0.f, 0.f, 0.f, 0.f, 0.f};
      for (int n = 0; n < N_DIM; ++n) {
        float aw = s_aw[n * B_DIM + b];
        v8h v8 = *(const v8h*)(vp + (size_t)n * B_DIM * H_DIM);
#pragma unroll
        for (int i = 0; i < 8; ++i) acc[i] += aw * (float)v8[i];
      }
      float inv = 1.f / s_sw[b];
#pragma unroll
      for (int i = 0; i < 8; ++i)
        x16[b * H_DIM + cg * 8 + i] = (_Float16)(acc[i] * inv);
    }
    __syncthreads();

    // --- E) gates = x@Wih^T + h@Whh^T : wave w owns 4 N-tiles ---
#pragma unroll
    for (int j = 0; j < 4; ++j) {
      int nb = (wave * 4 + j) * 16;
      v8f g = {};
      for (int kb = 0; kb < C_DIM; kb += 32)
        g = wmma16(afrag_lds(x16, C_DIM, kb, lane),
                   bfrag_g(Wih, C_DIM, nb, kb, lane), g);
      for (int kb = 0; kb < H_DIM; kb += 32)
        g = wmma16(afrag_lds(h16, H_DIM, kb, lane),
                   bfrag_g(Whh, H_DIM, nb, kb, lane), g);
      int nn = nb + (lane & 15);
      int mr = (lane >> 4) * 8;
#pragma unroll
      for (int i = 0; i < 8; ++i) gbuf[(mr + i) * H4_DIM + nn] = g[i];
    }
    __threadfence_block();
    __syncthreads();

    // --- F) LSTM cell, update h16 / cpriv, emit hs ---
#pragma unroll
    for (int j = 0; j < 8; ++j) {
      int idx = tid + j * 1024;
      int b = idx >> 9, hh = idx & 511;
      const float* gr = gbuf + b * H4_DIM;
      float gi = gr[hh]        + bih[hh]        + bhh[hh];
      float gf = gr[512 + hh]  + bih[512 + hh]  + bhh[512 + hh];
      float gg = gr[1024 + hh] + bih[1024 + hh] + bhh[1024 + hh];
      float go = gr[1536 + hh] + bih[1536 + hh] + bhh[1536 + hh];
      float cn = sigmoidf(gf) * cpriv[j] + sigmoidf(gi) * tanhf(gg);
      float hn = sigmoidf(go) * tanhf(cn);
      cpriv[j] = cn;
      h16[idx] = (_Float16)hn;
      hs[((size_t)l * B_DIM + b) * H_DIM + hh] = hn;
      if (l == L_DIM - 1) { outh[idx] = hn; outc[idx] = cn; }
    }
    __syncthreads();
  }
}

// ----------------------------------------------------- logits + softmax (f32)

__global__ __launch_bounds__(64) void k3_logits(
    const float* __restrict__ hs, const float* __restrict__ fcW,
    const float* __restrict__ fcb, float* __restrict__ probs) {
  __shared__ float sl[O_DIM];
  __shared__ float sred[2];
  int lb = blockIdx.x;
  int l = lb >> 4, b = lb & 15;
  int o = threadIdx.x;
  if (o < O_DIM) {
    const float* hp = hs + ((size_t)l * B_DIM + b) * H_DIM;
    const float* wp = fcW + (size_t)o * H_DIM;
    float acc = fcb[o];
    for (int k = 0; k < H_DIM; k += 4)
      acc += hp[k] * wp[k] + hp[k + 1] * wp[k + 1] + hp[k + 2] * wp[k + 2] +
             hp[k + 3] * wp[k + 3];
    sl[o] = acc;
  }
  __syncthreads();
  if (threadIdx.x == 0) {
    float mx = -1e30f;
    for (int i = 0; i < O_DIM; ++i) mx = fmaxf(mx, sl[i]);
    float sum = 0.f;
    for (int i = 0; i < O_DIM; ++i) sum += __expf(sl[i] - mx);
    sred[0] = mx; sred[1] = sum;
  }
  __syncthreads();
  if (o < O_DIM)
    probs[((size_t)b * L_DIM + l) * O_DIM + o] = __expf(sl[o] - sred[0]) / sred[1];
}

// --------------------------------------------------------------- launch

extern "C" void kernel_launch(void* const* d_in, const int* in_sizes, int n_in,
                              void* d_out, int out_size, void* d_ws,
                              size_t ws_size, hipStream_t stream) {
  const float* feat  = (const float*)d_in[0];
  const float* prior = (const float*)d_in[1];
  const float* h0h   = (const float*)d_in[2];
  const float* h0c   = (const float*)d_in[3];
  const float* v     = (const float*)d_in[4];
  const float* Wh    = (const float*)d_in[5];
  const float* Wk    = (const float*)d_in[6];
  const float* Wv    = (const float*)d_in[7];
  const float* Wih   = (const float*)d_in[8];
  const float* Whh   = (const float*)d_in[9];
  const float* bih   = (const float*)d_in[10];
  const float* bhh   = (const float*)d_in[11];
  const float* fcW   = (const float*)d_in[12];
  const float* fcb   = (const float*)d_in[13];
  float* out = (float*)d_out;

  char* wp = (char*)d_ws;
  auto alloc = [&](size_t bytes) -> char* {
    char* p = wp;
    wp += (bytes + 255) & ~(size_t)255;
    return p;
  };
  _Float16* WkT16 = (_Float16*)alloc((size_t)512 * 512 * 2);
  _Float16* WvT16 = (_Float16*)alloc((size_t)512 * 512 * 2);
  _Float16* WhT16 = (_Float16*)alloc((size_t)512 * 512 * 2);
  _Float16* Wih16 = (_Float16*)alloc((size_t)2048 * 512 * 2);
  _Float16* Whh16 = (_Float16*)alloc((size_t)2048 * 512 * 2);
  _Float16* key16 = (_Float16*)alloc((size_t)L_DIM * N_DIM * B_DIM * H_DIM * 2);
  _Float16* val16 = (_Float16*)alloc((size_t)L_DIM * N_DIM * B_DIM * H_DIM * 2);
  float* hs   = (float*)alloc((size_t)L_DIM * B_DIM * H_DIM * 4);
  float* qbuf = (float*)alloc((size_t)B_DIM * H_DIM * 4);
  float* gbuf = (float*)alloc((size_t)B_DIM * H4_DIM * 4);

  k0_transpose<<<(512 * 512 + 255) / 256, 256, 0, stream>>>(Wk, WkT16, 512, 512);
  k0_transpose<<<(512 * 512 + 255) / 256, 256, 0, stream>>>(Wv, WvT16, 512, 512);
  k0_transpose<<<(512 * 512 + 255) / 256, 256, 0, stream>>>(Wh, WhT16, 512, 512);
  k0_convert<<<(2048 * 512 + 255) / 256, 256, 0, stream>>>(Wih, Wih16, 2048 * 512);
  k0_convert<<<(2048 * 512 + 255) / 256, 256, 0, stream>>>(Whh, Whh16, 2048 * 512);

  k1_keyval<<<(L_DIM * N_DIM * B_DIM) / 16, 512, 0, stream>>>(feat, WkT16, WvT16,
                                                              key16, val16);

  k2_scan<<<1, 1024, 0, stream>>>(key16, val16, prior, h0h, h0c, v, WhT16,
                                  Wih16, Whh16, bih, bhh, hs, qbuf, gbuf,
                                  out + B_DIM * L_DIM * O_DIM,
                                  out + B_DIM * L_DIM * O_DIM + B_DIM * H_DIM);

  k3_logits<<<L_DIM * B_DIM, 64, 0, stream>>>(hs, fcW, fcb, out);

  (void)in_sizes; (void)n_in; (void)out_size; (void)ws_size;
}
